// GNN_29197187678384
// MI455X (gfx1250) — matrix operand order repaired and checked by reading, not compiled, not run on hardware
//
#include <hip/hip_runtime.h>
#include <hip/hip_fp16.h>

typedef __attribute__((ext_vector_type(16))) _Float16 v16h;
typedef __attribute__((ext_vector_type(8)))  float    v8f;

__device__ __forceinline__ void atomAddF(float* p, float v) {
  // Lowers to global_atomic_add_f32 (no CAS loop) at agent scope.
  __hip_atomic_fetch_add(p, v, __ATOMIC_RELAXED, __HIP_MEMORY_SCOPE_AGENT);
}

// ---- init: deg=1 (self loop), zero layer-1 accumulator and output accumulator
__global__ void gcn_init(float* __restrict__ deg, float* __restrict__ agg1,
                         float* __restrict__ out2, int N) {
  long long i = (long long)blockIdx.x * blockDim.x + threadIdx.x;
  long long tot = (long long)N * 64;
  if (i < tot) agg1[i] = 0.0f;
  if (i < N) { deg[i] = 1.0f; out2[i] = 0.0f; }
}

// ---- degree over destinations (row)
__global__ void gcn_degree(const int* __restrict__ rowv, float* __restrict__ deg, int E) {
  int e = blockIdx.x * blockDim.x + threadIdx.x;
  if (e < E) atomAddF(&deg[rowv[e]], 1.0f);
}

// ---- dinv = rsqrt(deg) in place (deg >= 1 always due to self loops)
__global__ void gcn_dinv(float* __restrict__ deg, int N) {
  int i = blockIdx.x * blockDim.x + threadIdx.x;
  if (i < N) { float d = deg[i]; deg[i] = (d > 0.0f) ? rsqrtf(d) : 0.0f; }
}

// ---- norm[e] = dinv[row]*dinv[col]; last N entries are self loops
__global__ void gcn_norm(const int* __restrict__ rowv, const int* __restrict__ colv,
                         const float* __restrict__ dinv, float* __restrict__ norm,
                         int E, int N) {
  int e = blockIdx.x * blockDim.x + threadIdx.x;
  if (e >= E + N) return;
  int r, c;
  if (e < E) { r = rowv[e]; c = colv[e]; } else { r = c = e - E; }
  norm[e] = dinv[r] * dinv[c];
}

// ---- h1 = x @ W1 via v_wmma_f32_16x16x32_f16.
//      W1 staged once per block into LDS as f16, transposed to [n][k] so each
//      B fragment is two aligned 16B LDS reads (ds_load_b128 x2, no cvt).
//      One wave per 16-row tile covering all 4 col-tiles (A reuse); K=128=4x32.
__global__ void gcn_gemm1_wmma(const float* __restrict__ x,
                               const float* __restrict__ W1,
                               float* __restrict__ h1, int mTiles) {
  __shared__ __align__(32) _Float16 w1t[64 * 128];   // [n][k], 16 KB

  // cooperative stage + transpose + f32->f16 (float4 global reads)
  {
    const float4* w4 = reinterpret_cast<const float4*>(W1);   // 2048 float4s
    for (int f = threadIdx.x; f < 2048; f += 256) {
      int k  = f >> 4;              // 16 float4 per k-row (64 cols)
      int nb = (f & 15) * 4;
      float4 v = w4[f];
      w1t[(nb + 0) * 128 + k] = (_Float16)v.x;
      w1t[(nb + 1) * 128 + k] = (_Float16)v.y;
      w1t[(nb + 2) * 128 + k] = (_Float16)v.z;
      w1t[(nb + 3) * 128 + k] = (_Float16)v.w;
    }
  }
  __syncthreads();

  int wave = threadIdx.x >> 5;
  int lane = threadIdx.x & 31;
  int mTile = blockIdx.x * 8 + wave;          // blockDim = 256 -> 8 waves
  if (mTile >= mTiles) return;                // wave-uniform: EXEC stays all-1s
  int m = lane & 15;                          // A row / B col / D col
  int g = lane >> 4;                          // lane half-group

  v8f acc[4] = {v8f{}, v8f{}, v8f{}, v8f{}};

  for (int kk = 0; kk < 4; ++kk) {
    int kb = kk * 32;
    // A fragment: 16-bit A 16x32 layout (ISA 7.12.2); float4 -> b128 loads
    const float* xr = x + (long long)(mTile * 16 + m) * 128 + kb + g * 8;
    float4 p0 = reinterpret_cast<const float4*>(xr)[0];
    float4 p1 = reinterpret_cast<const float4*>(xr)[1];        // K = kb+g*8+{0..7}
    float4 q0 = reinterpret_cast<const float4*>(xr + 16)[0];
    float4 q1 = reinterpret_cast<const float4*>(xr + 16)[1];   // K = kb+16+g*8+{0..7}
    v16h a;
    a[0]  = (_Float16)p0.x; a[1]  = (_Float16)p0.y; a[2]  = (_Float16)p0.z; a[3]  = (_Float16)p0.w;
    a[4]  = (_Float16)p1.x; a[5]  = (_Float16)p1.y; a[6]  = (_Float16)p1.z; a[7]  = (_Float16)p1.w;
    a[8]  = (_Float16)q0.x; a[9]  = (_Float16)q0.y; a[10] = (_Float16)q0.z; a[11] = (_Float16)q0.w;
    a[12] = (_Float16)q1.x; a[13] = (_Float16)q1.y; a[14] = (_Float16)q1.z; a[15] = (_Float16)q1.w;
#pragma unroll
    for (int nt = 0; nt < 4; ++nt) {
      // B fragment: col = m, halves j<8 -> K = kb + g*8 + j, j>=8 -> K = kb + 16 + g*8 + (j-8)
      const _Float16* bp = &w1t[(nt * 16 + m) * 128 + kb + g * 8];
      v16h b;
#pragma unroll
      for (int j = 0; j < 8; ++j) { b[j] = bp[j]; b[j + 8] = bp[16 + j]; }
      acc[nt] = __builtin_amdgcn_wmma_f32_16x16x32_f16(
          false, a, false, b, (short)0, acc[nt], false, false);
    }
  }
  // D layout: VGPR r -> M = r + 8*g, col = m
#pragma unroll
  for (int nt = 0; nt < 4; ++nt) {
#pragma unroll
    for (int r = 0; r < 8; ++r) {
      int orow = mTile * 16 + r + 8 * g;
      h1[(long long)orow * 64 + nt * 16 + m] = acc[nt][r];
    }
  }
}

// ---- layer-1 propagate: one wave per edge; lane handles 2 features (float2)
__global__ void gcn_scatter1(const int* __restrict__ rowv, const int* __restrict__ colv,
                             const float* __restrict__ norm, const float* __restrict__ h1,
                             float* __restrict__ agg1, int E, int N) {
  long long t = (long long)blockIdx.x * blockDim.x + threadIdx.x;
  long long we = t >> 5;
  int lane = (int)(t & 31);
  if (we >= (long long)E + N) return;
  int r, c;
  if (we < E) { r = rowv[we]; c = colv[we]; } else { r = c = (int)(we - E); }
  float nr = norm[we];
  float2 v = reinterpret_cast<const float2*>(h1 + (long long)c * 64)[lane];
  float* ap = agg1 + (long long)r * 64 + 2 * lane;
  atomAddF(ap,     v.x * nr);
  atomAddF(ap + 1, v.y * nr);
}

// ---- bias + ReLU + matvec with W2 (64 -> 1): wave per node, shuffle reduce
__global__ void gcn_layer2(const float* __restrict__ agg1, const float* __restrict__ b1,
                           const float* __restrict__ W2, float* __restrict__ h2, int N) {
  long long t = (long long)blockIdx.x * blockDim.x + threadIdx.x;
  long long n = t >> 5;
  int lane = (int)(t & 31);
  if (n >= N) return;
  const float* hv = agg1 + n * 64;
  float v = fmaxf(hv[lane] + b1[lane], 0.0f) * W2[lane] +
            fmaxf(hv[lane + 32] + b1[lane + 32], 0.0f) * W2[lane + 32];
#pragma unroll
  for (int off = 16; off > 0; off >>= 1) v += __shfl_xor(v, off, 32);
  if (lane == 0) h2[n] = v;
}

// ---- layer-2 propagate: scalar per edge
__global__ void gcn_scatter2(const int* __restrict__ rowv, const int* __restrict__ colv,
                             const float* __restrict__ norm, const float* __restrict__ h2,
                             float* __restrict__ out2, int E, int N) {
  int e = blockIdx.x * blockDim.x + threadIdx.x;
  if (e >= E + N) return;
  int r, c;
  if (e < E) { r = rowv[e]; c = colv[e]; } else { r = c = e - E; }
  atomAddF(&out2[r], h2[c] * norm[e]);
}

// ---- +b2 and sigmoid, in place on d_out
__global__ void gcn_final(float* __restrict__ out2, const float* __restrict__ b2, int N) {
  int i = blockIdx.x * blockDim.x + threadIdx.x;
  if (i < N) {
    float z = out2[i] + b2[0];
    out2[i] = 1.0f / (1.0f + __expf(-z));
  }
}

extern "C" void kernel_launch(void* const* d_in, const int* in_sizes, int n_in,
                              void* d_out, int out_size, void* d_ws, size_t ws_size,
                              hipStream_t stream) {
  const float* x   = (const float*)d_in[0];
  const int*   ei  = (const int*)d_in[1];      // [2, E] flattened
  const float* W1  = (const float*)d_in[2];
  const float* b1  = (const float*)d_in[3];
  const float* W2  = (const float*)d_in[4];
  const float* b2  = (const float*)d_in[5];
  float* out = (float*)d_out;

  const int N = in_sizes[0] / 128;
  const int E = in_sizes[1] / 2;
  const int* rowv = ei;          // destinations
  const int* colv = ei + E;      // sources

  // workspace layout (floats)
  float* ws   = (float*)d_ws;
  float* deg  = ws;                              // N      (becomes dinv)
  float* norm = deg + N;                         // E + N
  float* h1   = norm + (size_t)E + N;            // N*64   (x @ W1)
  float* agg1 = h1 + (size_t)N * 64;             // N*64   (propagated layer-1)
  float* h2   = agg1 + (size_t)N * 64;           // N      (layer-2 pre-propagate)

  const int B = 256;
  long long totInit = (long long)N * 64;
  int mTiles = (N + 15) / 16;
  long long sc1Threads = ((long long)E + N) * 32;

  gcn_init<<<(int)((totInit + B - 1) / B), B, 0, stream>>>(deg, agg1, out, N);
  gcn_degree<<<(E + B - 1) / B, B, 0, stream>>>(rowv, deg, E);
  gcn_dinv<<<(N + B - 1) / B, B, 0, stream>>>(deg, N);
  gcn_norm<<<(E + N + B - 1) / B, B, 0, stream>>>(rowv, colv, deg, norm, E, N);
  gcn_gemm1_wmma<<<(mTiles + 7) / 8, B, 0, stream>>>(x, W1, h1, mTiles);
  gcn_scatter1<<<(int)((sc1Threads + B - 1) / B), B, 0, stream>>>(rowv, colv, norm,
                                                                  h1, agg1, E, N);
  gcn_layer2<<<(int)(((long long)N * 32 + B - 1) / B), B, 0, stream>>>(agg1, b1, W2, h2, N);
  gcn_scatter2<<<(E + N + B - 1) / B, B, 0, stream>>>(rowv, colv, norm, h2, out, E, N);
  gcn_final<<<(N + B - 1) / B, B, 0, stream>>>(out, b2, N);
}
